// DisentangledSpatialAttention_73564199845902
// MI455X (gfx1250) — matrix operand order repaired and checked by reading, not compiled, not run on hardware
//
#include <hip/hip_runtime.h>
#include <hip/hip_bf16.h>
#include <math.h>

typedef __attribute__((ext_vector_type(16))) __bf16 v16bf;
typedef __attribute__((ext_vector_type(8)))  __bf16 v8bf;
typedef __attribute__((ext_vector_type(8)))  float  v8f;

#define NB 4
#define NS 2048
#define NE 1024
#define NH 16
#define ND 64
#define NROWS (NB*NS)      /* 8192 */
#define KCAT  (2*NE)       /* 2048 */
#define NCAT  (5*NE)       /* 5120 */

#define WMMA_BF16(a,b,c) \
  __builtin_amdgcn_wmma_f32_16x16x32_bf16(false,(a),false,(b),(short)0,(c),false,false)

// LDS byte offset of a __shared__ object: generic -> AS(3) -> int.
static __device__ __forceinline__ unsigned lds_off(const void* p) {
  return (unsigned)(unsigned long long)
      (__attribute__((address_space(3))) const char*)p;
}
// Async memory->LDS 16B copy (per lane). Tracked by ASYNCcnt; no VGPR staging.
#define ASYNC_LD_B128(ldsp, gp)                                        \
  asm volatile("global_load_async_to_lds_b128 %0, %1, off"             \
               :: "v"(lds_off(ldsp)), "v"(gp) : "memory")
#define WAIT_ASYNC() asm volatile("s_wait_asynccnt 0x0" ::: "memory")

// A-operand fragment: per ISA 7.12.2, lanes 0-15 hold K {0..7,16..23},
// lanes 16-31 hold K {8..15,24..31}. Caller pre-offsets by khalf*8.
static __device__ __forceinline__ v16bf ldA(const __bf16* p) {
  v16bf a;
  ((v8bf*)&a)[0] = *(const v8bf*)(p);
  ((v8bf*)&a)[1] = *(const v8bf*)(p + 16);
  return a;
}
// B-operand fragment: lanes 0-15 hold K {0..7,16..23}, lanes 16-31 hold
// K {8..15,24..31} of their column; caller pre-offsets by khalf*8.
static __device__ __forceinline__ v16bf ldB(const __bf16* p) {
  v16bf b;
  ((v8bf*)&b)[0] = *(const v8bf*)(p);
  ((v8bf*)&b)[1] = *(const v8bf*)(p + 16);
  return b;
}

// ---------------------------------------------------------------- prep ----
__global__ __launch_bounds__(256)
void prep_inputs(const float* __restrict__ x, __bf16* __restrict__ xcat,
                 float* __restrict__ out) {
  size_t idx = (size_t)blockIdx.x * 256 + threadIdx.x;  // < NROWS*NE
  int    e   = (int)(idx & (NE - 1));
  size_t r   = idx >> 10;
  float vt = x[(r * 2) * NE + e];
  float vs = x[(r * 2 + 1) * NE + e];
  xcat[r * KCAT + e]      = (__bf16)vt;
  xcat[r * KCAT + NE + e] = (__bf16)vs;
  out[(r * 2 + 1) * NE + e] = vs;  // pass-through channel
}

// Build Wcat^T (NCAT x KCAT, n-major) with lambdas folded in, plus Wo^T.
__global__ __launch_bounds__(256)
void prep_weights(const float* __restrict__ Wt, const float* __restrict__ Ws,
                  const float* __restrict__ Wo,
                  const float* __restrict__ lts_p, const float* __restrict__ lst_p,
                  const float* __restrict__ lss_p,
                  __bf16* __restrict__ WcatT, __bf16* __restrict__ WoT) {
  size_t idx = (size_t)blockIdx.x * 256 + threadIdx.x;
  const size_t NWC = (size_t)NCAT * KCAT;
  if (idx < NWC) {
    int    k  = (int)(idx & (KCAT - 1));
    size_t n  = idx >> 11;
    int seg = (int)(n >> 10);
    int nn  = (int)(n & 1023);
    bool lo = (k < NE);
    int  kk = lo ? k : k - NE;
    float lts = lts_p[0], lst = lst_p[0], lss = lss_p[0];
    float v;
    switch (seg) {
      case 0:  v = lo ? Wt[(size_t)kk * 3072 + nn]               : 0.0f; break;
      case 1:  v = lo ? 0.0f : Ws[(size_t)kk * 2048 + nn];               break;
      case 2:  v = lo ? Wt[(size_t)kk * 3072 + 1024 + nn]
                      : lts * Ws[(size_t)kk * 2048 + 1024 + nn];         break;
      case 3:  v = lo ? lst * Wt[(size_t)kk * 3072 + 1024 + nn]
                      : lss * Ws[(size_t)kk * 2048 + 1024 + nn];         break;
      default: v = lo ? Wt[(size_t)kk * 3072 + 2048 + nn]        : 0.0f; break;
    }
    WcatT[n * KCAT + k] = (__bf16)v;
  } else {
    size_t t = idx - NWC;
    if (t < (size_t)NE * NE) {
      int k = (int)(t & (NE - 1));
      int n = (int)(t >> 10);
      WoT[(size_t)n * NE + k] = (__bf16)Wo[(size_t)k * NE + n];
    }
  }
}

// ---------------------------------------------------------------- GEMM ----
// C(MxN) = A(MxK) * Bt(NxK)^T, bf16 in / f32 accumulate, WMMA 16x16x32.
// Block: 128x128 tile, 256 threads = 8 waves in 4(M) x 2(N), wave: 32x64.
// Tiles staged with GLOBAL_LOAD_ASYNC_TO_LDS (ASYNCcnt), double-buffered.
template <int EPI>
__global__ __launch_bounds__(256)
void gemm_wmma(const __bf16* __restrict__ A, const __bf16* __restrict__ Bt,
               int M, int N, int K, float* __restrict__ outF,
               __bf16* __restrict__ Qc, __bf16* __restrict__ Kc,
               __bf16* __restrict__ Vb) {
  __shared__ __align__(16) __bf16 As[2][128][40];  // 32 K + 8 pad
  __shared__ __align__(16) __bf16 Bs[2][128][40];

  const int tid   = threadIdx.x;
  const int lane  = tid & 31, wave = tid >> 5;
  const int khalf = lane >> 4, l16 = lane & 15;
  const int wm = wave & 3, wn = wave >> 2;
  const int rowBase = blockIdx.y * 128;
  const int colBase = blockIdx.x * 128;

  v8f acc[2][4];
#pragma unroll
  for (int i = 0; i < 2; i++)
#pragma unroll
    for (int j = 0; j < 4; j++) acc[i][j] = (v8f){0, 0, 0, 0, 0, 0, 0, 0};

  auto loadTile = [&](int buf, int kt) {
    const int kbase = kt * 32;
#pragma unroll
    for (int j = 0; j < 2; j++) {
      int c   = tid + 256 * j;  // 0..511 -> 16B chunks of an 8KB tile
      int row = c >> 2;
      int ko  = (c & 3) * 8;
      const __bf16* ga = A  + (size_t)(rowBase + row) * K + kbase + ko;
      const __bf16* gb = Bt + (size_t)(colBase + row) * K + kbase + ko;
      ASYNC_LD_B128(&As[buf][row][ko], ga);  // mem -> LDS, no VGPR staging
      ASYNC_LD_B128(&Bs[buf][row][ko], gb);
      __builtin_prefetch(ga + 64, 0, 1);  // global_prefetch_b8, 2 tiles ahead
      __builtin_prefetch(gb + 64, 0, 1);
    }
  };

  const int nk = K >> 5;
  loadTile(0, 0);
  WAIT_ASYNC();
  __syncthreads();
  int buf = 0;
  for (int kt = 0; kt < nk; ++kt) {
    if (kt + 1 < nk) loadTile(buf ^ 1, kt + 1);
    v16bf af[2], bfr[4];
#pragma unroll
    for (int mi = 0; mi < 2; mi++)
      af[mi] = ldA(&As[buf][wm * 32 + mi * 16 + l16][khalf * 8]);
#pragma unroll
    for (int ni = 0; ni < 4; ni++)
      bfr[ni] = ldB(&Bs[buf][wn * 64 + ni * 16 + l16][khalf * 8]);
#pragma unroll
    for (int mi = 0; mi < 2; mi++)
#pragma unroll
      for (int ni = 0; ni < 4; ni++)
        acc[mi][ni] = WMMA_BF16(af[mi], bfr[ni], acc[mi][ni]);
    WAIT_ASYNC();   // next tile fully in LDS before the barrier releases it
    __syncthreads();
    buf ^= 1;
  }

  // C layout: VGPR i holds row i (lanes 0-15) / i+8 (lanes 16-31), col = l16.
#pragma unroll
  for (int mi = 0; mi < 2; mi++)
#pragma unroll
    for (int ni = 0; ni < 4; ni++)
#pragma unroll
      for (int i = 0; i < 8; i++) {
        int r = rowBase + wm * 32 + mi * 16 + i + khalf * 8;
        int n = colBase + wn * 64 + ni * 16 + l16;
        float val = acc[mi][ni][i];
        if (EPI == 1) {
          outF[(size_t)r * (2 * NE) + n] = val;  // out[:,:,0,:]
        } else {
          int b = r >> 11, s = r & (NS - 1);
          int seg = n >> 10, jj = n & 1023, h = jj >> 6, d = jj & 63;
          size_t bhs = ((size_t)(b * NH + h) * NS + s);
          __bf16 bv = (__bf16)val;
          if (seg == 0)      Qc[bhs * 128 + d]      = bv;
          else if (seg == 1) Qc[bhs * 128 + 64 + d] = bv;
          else if (seg == 2) Kc[bhs * 128 + d]      = bv;
          else if (seg == 3) Kc[bhs * 128 + 64 + d] = bv;
          else               Vb[bhs * 64 + d]       = bv;
        }
      }
}

// ------------------------------------------------------- flash attention --
// grid (S/64, B*H), 128 threads = 4 waves; wave owns 16 query rows.
// Q,K are 128-dim (concatenated), V is 64-dim. Online softmax.
__global__ __launch_bounds__(128)
void flash_attn(const __bf16* __restrict__ Qc, const __bf16* __restrict__ Kc,
                const __bf16* __restrict__ Vb, __bf16* __restrict__ Y) {
  __shared__ __align__(16) __bf16 Ks[64][136];    // keys x dims (+pad)
  __shared__ __align__(16) __bf16 Vt[64][72];     // dims x keys (transposed)
  __shared__ __align__(16) __bf16 Ps[4][16][72];  // per-wave P bounce

  const int tid = threadIdx.x, lane = tid & 31, wave = tid >> 5;
  const int khalf = lane >> 4, l16 = lane & 15;
  const int bh = blockIdx.y;                 // b*NH + h
  const int qbase = blockIdx.x * 64 + wave * 16;
  const size_t bhRow = (size_t)bh * NS;

  // Q fragments: 16 rows x 128 dims = 4 K-chunks of 32
  v16bf qf[4];
  {
    const __bf16* qp = Qc + (bhRow + qbase + l16) * 128;
#pragma unroll
    for (int c = 0; c < 4; c++) qf[c] = ldA(qp + c * 32 + khalf * 8);
  }

  float m[8], l[8];
#pragma unroll
  for (int i = 0; i < 8; i++) { m[i] = -1e30f; l[i] = 0.0f; }
  v8f o[4];
#pragma unroll
  for (int i = 0; i < 4; i++) o[i] = (v8f){0, 0, 0, 0, 0, 0, 0, 0};

  for (int kb = 0; kb < NS / 64; ++kb) {
    // stage K block via async mem->LDS (64x128 bf16 = 8 x b128 per thread)
#pragma unroll
    for (int j = 0; j < 8; j++) {
      int c = tid + 128 * j;  // 1024 chunks of 8 elems
      int key = c >> 4;
      int off = (c & 15) * 8;
      ASYNC_LD_B128(&Ks[key][off],
                    Kc + (bhRow + kb * 64 + key) * 128 + off);
    }
    // stage V transposed (element-wise, through VGPRs)
    {
      int key = tid >> 1;
      int dh  = (tid & 1) * 32;
      const __bf16* vp = Vb + (bhRow + kb * 64 + key) * 64 + dh;
#pragma unroll
      for (int d = 0; d < 32; d++) Vt[dh + d][key] = vp[d];
    }
    WAIT_ASYNC();
    __syncthreads();

    // scores S = Q K^T : 16x64 f32, 4 chained WMMAs per 16-key tile
    v8f sc[4];
#pragma unroll
    for (int ni = 0; ni < 4; ni++) {
      sc[ni] = (v8f){0, 0, 0, 0, 0, 0, 0, 0};
#pragma unroll
      for (int c = 0; c < 4; c++) {
        v16bf bfr = ldB(&Ks[ni * 16 + l16][c * 32 + khalf * 8]);
        sc[ni] = WMMA_BF16(qf[c], bfr, sc[ni]);
      }
    }
#pragma unroll
    for (int ni = 0; ni < 4; ni++)
#pragma unroll
      for (int i = 0; i < 8; i++) sc[ni][i] *= 0.125f;  // 1/sqrt(64)

    // online softmax: row reductions across the 16-lane half owning each row
    float fsc[8], rsum[8];
#pragma unroll
    for (int i = 0; i < 8; i++) {
      float v0 = fmaxf(fmaxf(sc[0][i], sc[1][i]), fmaxf(sc[2][i], sc[3][i]));
#pragma unroll
      for (int msk = 1; msk < 16; msk <<= 1) v0 = fmaxf(v0, __shfl_xor(v0, msk, 32));
      float mn = fmaxf(m[i], v0);
      fsc[i] = __expf(m[i] - mn);
      m[i] = mn;
      rsum[i] = 0.0f;
    }
#pragma unroll
    for (int ni = 0; ni < 4; ni++)
#pragma unroll
      for (int i = 0; i < 8; i++) {
        float p = __expf(sc[ni][i] - m[i]);
        rsum[i] += p;
        Ps[wave][i + khalf * 8][ni * 16 + l16] = (__bf16)p;  // C -> A reshape
      }
#pragma unroll
    for (int i = 0; i < 8; i++) {
      float s0 = rsum[i];
#pragma unroll
      for (int msk = 1; msk < 16; msk <<= 1) s0 += __shfl_xor(s0, msk, 32);
      l[i] = l[i] * fsc[i] + s0;
    }
#pragma unroll
    for (int ni = 0; ni < 4; ni++)
#pragma unroll
      for (int i = 0; i < 8; i++) o[ni][i] *= fsc[i];

    // O += P(16x64) * V(64x64); same-wave DS ops are in order (no barrier)
#pragma unroll
    for (int c = 0; c < 2; c++) {
      v16bf pf = ldA(&Ps[wave][l16][c * 32 + khalf * 8]);
#pragma unroll
      for (int ni = 0; ni < 4; ni++) {
        v16bf vf = ldB(&Vt[ni * 16 + l16][c * 32 + khalf * 8]);
        o[ni] = WMMA_BF16(pf, vf, o[ni]);
      }
    }
    __syncthreads();
  }

  const int b = bh >> 4, h = bh & 15;
#pragma unroll
  for (int ni = 0; ni < 4; ni++)
#pragma unroll
    for (int i = 0; i < 8; i++) {
      int srow = qbase + i + khalf * 8;
      float val = o[ni][i] / l[i];
      Y[((size_t)b * NS + srow) * NE + h * 64 + ni * 16 + l16] = (__bf16)val;
    }
}

// -------------------------------------------------------------- launch ----
extern "C" void kernel_launch(void* const* d_in, const int* in_sizes, int n_in,
                              void* d_out, int out_size, void* d_ws, size_t ws_size,
                              hipStream_t stream) {
  (void)in_sizes; (void)n_in; (void)out_size; (void)ws_size;
  const float* x      = (const float*)d_in[0];
  const float* Wt     = (const float*)d_in[1];
  const float* Ws     = (const float*)d_in[2];
  const float* Wo     = (const float*)d_in[3];
  const float* lam_ts = (const float*)d_in[4];
  const float* lam_st = (const float*)d_in[5];
  const float* lam_ss = (const float*)d_in[6];
  float* out = (float*)d_out;

  char* ws = (char*)d_ws;
  size_t off = 0;
  auto alloc = [&](size_t bytes) -> void* {
    void* p = ws + off;
    off += (bytes + 255) & ~(size_t)255;
    return p;
  };
  __bf16* xcat  = (__bf16*)alloc((size_t)NROWS * KCAT * 2);      // 32 MB
  __bf16* WcatT = (__bf16*)alloc((size_t)NCAT * KCAT * 2);       // 20 MB
  __bf16* WoT   = (__bf16*)alloc((size_t)NE * NE * 2);           //  2 MB
  __bf16* Qc    = (__bf16*)alloc((size_t)NB * NH * NS * 128 * 2);// 32 MB
  __bf16* Kc    = (__bf16*)alloc((size_t)NB * NH * NS * 128 * 2);// 32 MB
  __bf16* Vb    = (__bf16*)alloc((size_t)NB * NH * NS * 64 * 2); // 16 MB
  __bf16* Yb    = (__bf16*)alloc((size_t)NROWS * NE * 2);        // 16 MB

  // 1) fp32 -> bf16, build [xt|xs]; copy xs to out channel 1
  prep_inputs<<<(NROWS * NE) / 256, 256, 0, stream>>>(x, xcat, out);

  // 2) pre-mixed transposed weights (lambdas folded in)
  {
    size_t total = (size_t)NCAT * KCAT + (size_t)NE * NE;
    prep_weights<<<(unsigned)((total + 255) / 256), 256, 0, stream>>>(
        Wt, Ws, Wo, lam_ts, lam_st, lam_ss, WcatT, WoT);
  }

  // 3) fused QKV projection: (8192x2048) * (2048x5120)
  gemm_wmma<0><<<dim3(NCAT / 128, NROWS / 128), 256, 0, stream>>>(
      xcat, WcatT, NROWS, NCAT, KCAT, nullptr, Qc, Kc, Vb);

  // 4) attention (head-dim 128 logits, 64-dim values)
  flash_attn<<<dim3(NS / 64, NB * NH), 128, 0, stream>>>(Qc, Kc, Vb, Yb);

  // 5) output projection -> out channel 0 (fp32)
  gemm_wmma<1><<<dim3(NE / 128, NROWS / 128), 256, 0, stream>>>(
      Yb, WoT, NROWS, NE, NE, out, nullptr, nullptr, nullptr);
}
